// RNNCell_Encoder_81535659147676
// MI455X (gfx1250) — compile-verified, matrix operand order checked by reading, not run on hardware
//
#include <hip/hip_runtime.h>
#include <hip/hip_bf16.h>
#include <math.h>

typedef __attribute__((ext_vector_type(16))) __bf16 v16bf;
typedef __attribute__((ext_vector_type(8)))  __bf16 v8bf;
typedef __attribute__((ext_vector_type(8)))  float  v8f;

#define SEQ   1024
#define BATCH 256
#define DIM   512
#define HID   512
#define NWG2  16   // persistent workgroups in the scan kernel (4x4 tile grid)

// ---- workspace layout (bytes) ----
static constexpr size_t OFF_WIH  = 0;                                   // 512x512 bf16 = 512KB
static constexpr size_t OFF_WHH  = OFF_WIH + (size_t)DIM * HID * 2;     // 512KB
static constexpr size_t OFF_BIAS = OFF_WHH + (size_t)DIM * HID * 2;     // 512 f32
static constexpr size_t OFF_H0   = OFF_BIAS + 4096;                     // 256x512 f32 = 512KB
static constexpr size_t OFF_H1   = OFF_H0 + (size_t)BATCH * HID * 4;    // 512KB
static constexpr size_t OFF_CNT  = OFF_H1 + (size_t)BATCH * HID * 4;    // barrier counter
static constexpr size_t OFF_XP   = OFF_CNT + 4096;                      // 1024x256x512 f32 = 512MB

// ---------------------------------------------------------------------------
// Fragment loaders. wave32 16-bit WMMA A/B fragment mapping (identical for A
// and B since our B rows (= W rows) are contiguous in K):
//   lane L owns matrix row/col (L & 15)
//   elements 0..7  <- K = kbase + ((L&16)?8:0) + 0..7
//   elements 8..15 <- K = kbase + ((L&16)?8:0) + 16..23
// ---------------------------------------------------------------------------
__device__ __forceinline__ v16bf load_frag_f32(const float* __restrict__ p) {
  const float4* q = (const float4*)p;
  float4 x0 = q[0];  // k+0..3
  float4 x1 = q[1];  // k+4..7
  float4 x2 = q[4];  // k+16..19
  float4 x3 = q[5];  // k+20..23
  v16bf f;
  f[0]  = (__bf16)x0.x; f[1]  = (__bf16)x0.y; f[2]  = (__bf16)x0.z; f[3]  = (__bf16)x0.w;
  f[4]  = (__bf16)x1.x; f[5]  = (__bf16)x1.y; f[6]  = (__bf16)x1.z; f[7]  = (__bf16)x1.w;
  f[8]  = (__bf16)x2.x; f[9]  = (__bf16)x2.y; f[10] = (__bf16)x2.z; f[11] = (__bf16)x2.w;
  f[12] = (__bf16)x3.x; f[13] = (__bf16)x3.y; f[14] = (__bf16)x3.z; f[15] = (__bf16)x3.w;
  return f;
}

__device__ __forceinline__ v16bf load_frag_bf16(const __bf16* __restrict__ p) {
  v8bf lo = *(const v8bf*)(p);       // k+0..7   (16B aligned: k offsets multiples of 8, ld=512)
  v8bf hi = *(const v8bf*)(p + 16);  // k+16..23
  v16bf f;
#pragma unroll
  for (int i = 0; i < 8; ++i) { f[i] = lo[i]; f[i + 8] = hi[i]; }
  return f;
}

// ---------------------------------------------------------------------------
// Kernel 0: prep — weights to bf16 (L2-resident thereafter), fused bias, h0=0,
// barrier counter = 0 (must be re-done every call: graph replays reuse ws).
// ---------------------------------------------------------------------------
__global__ __launch_bounds__(256) void rnn_init_kernel(
    const float* __restrict__ Wih, const float* __restrict__ Whh,
    const float* __restrict__ bih, const float* __restrict__ bhh,
    __bf16* __restrict__ wih_bf, __bf16* __restrict__ whh_bf,
    float* __restrict__ bias, float* __restrict__ h0, int* __restrict__ cnt) {
  int idx = blockIdx.x * 256 + threadIdx.x;
  if (idx < DIM * HID) {
    wih_bf[idx] = (__bf16)Wih[idx];
    whh_bf[idx] = (__bf16)Whh[idx];
  }
  if (idx < BATCH * HID) h0[idx] = 0.0f;
  if (idx < HID) bias[idx] = bih[idx] + bhh[idx];
  if (idx == 0) *cnt = 0;
}

// ---------------------------------------------------------------------------
// Kernel 1: x_proj = inputs @ W_ih^T + (b_ih + b_hh)   [262144 x 512]
// Per wave: 16M x 128N slab (8 f32 accumulators), K-loop of 16 (K=512, 32/step).
// Wave map: one WG = 2 m-tiles x 4 n-groups, so the 4 waves sharing an A tile
// are co-resident on one WGP -> A rows fetched ~once from HBM (512MB, ~22us),
// redundant fetches are WGP$/L2 hits. W_ih (bf16, 512KB) stays L2-resident.
// Nontemporal stores: x_proj (512MB) must not evict the L2 working set.
// ---------------------------------------------------------------------------
__global__ __launch_bounds__(256) void rnn_xproj_kernel(
    const float* __restrict__ x, const __bf16* __restrict__ wih_bf,
    const float* __restrict__ bias, float* __restrict__ xproj) {
  const int lane  = threadIdx.x & 31;
  const int wv    = threadIdx.x >> 5;
  const int mt    = blockIdx.x * 2 + (wv >> 2);   // 8192 blocks x 2 = 16384 m-tiles
  const int n0    = (wv & 3) * 128;               // 4 n-groups of 128 within the WG
  const int m0    = mt * 16;
  const int khalf = (lane & 16) ? 8 : 0;
  const int arow  = m0 + (lane & 15);
  const float* aptr = x + (size_t)arow * DIM + khalf;

  v8f acc[8] = {};
  for (int k0 = 0; k0 < DIM; k0 += 32) {
    __builtin_prefetch(aptr + k0 + 64, 0, 1);      // stream next A chunk (global_prefetch)
    v16bf a = load_frag_f32(aptr + k0);
#pragma unroll
    for (int j = 0; j < 8; ++j) {
      const __bf16* bp = wih_bf + (size_t)(n0 + j * 16 + (lane & 15)) * DIM + k0 + khalf;
      v16bf b = load_frag_bf16(bp);
      acc[j] = __builtin_amdgcn_wmma_f32_16x16x32_bf16(
          false, a, false, b, (short)0, acc[j], false, false);
    }
  }

  const int hi8 = (lane & 16) ? 8 : 0;
#pragma unroll
  for (int j = 0; j < 8; ++j) {
    const int col = n0 + j * 16 + (lane & 15);
    const float bv = bias[col];
#pragma unroll
    for (int r = 0; r < 8; ++r) {
      const int row = m0 + r + hi8;
      __builtin_nontemporal_store(acc[j][r] + bv, &xproj[(size_t)row * HID + col]);
    }
  }
}

// ---------------------------------------------------------------------------
// Kernel 2: persistent sequential scan. 16 WGs arranged 4(M) x 4(N); each WG
// owns a 64-row x 128-col slab of h. Its time-invariant W_hh slab
// (128 rows x 512 K, bf16 = 128KB) is staged into LDS ONCE before the
// 1024-step loop, then every step reads B fragments via ds_load — removing
// ~8GB of per-step L2 weight traffic from the latency-critical serial chain.
// Per wave (8 per WG, 2x4): 16M x 64N slab, 4 WMMA accumulators, K-loop of 16.
// h is double-buffered in ws; device-scope counter barrier per step.
// ---------------------------------------------------------------------------
__global__ __launch_bounds__(256) void rnn_scan_kernel(
    const float* __restrict__ xproj, const __bf16* __restrict__ whh_bf,
    float* __restrict__ h0buf, float* __restrict__ h1buf, int* __restrict__ cnt) {
  extern __shared__ char smem[];
  __bf16* wlds = (__bf16*)smem;                 // [128 rows][512 K] bf16 = 128KB

  const int lane = threadIdx.x & 31;
  const int wv   = threadIdx.x >> 5;            // 0..7
  const int mblk = blockIdx.x >> 2;             // 0..3 -> rows [mblk*64, +64)
  const int nblk = blockIdx.x & 3;              // 0..3 -> cols [nblk*128, +128)

  // ---- stage this WG's W_hh slab into LDS (once) ----
  {
    const v8bf* src = (const v8bf*)(whh_bf + (size_t)(nblk * 128) * HID);
    v8bf* dst = (v8bf*)wlds;
    for (int i = threadIdx.x; i < (128 * HID) / 8; i += 256)
      dst[i] = src[i];
  }
  __syncthreads();

  const int m0    = mblk * 64 + (wv >> 1) * 16;   // global row base of this wave
  const int n0l   = (wv & 1) * 64;                // local col base within LDS slab
  const int n0g   = nblk * 128 + n0l;             // global col base
  const int khalf = (lane & 16) ? 8 : 0;
  const int hi8   = khalf;
  const int arow  = m0 + (lane & 15);
  float* hb[2] = {h0buf, h1buf};

  for (int t = 0; t < SEQ; ++t) {
    const float* hin  = hb[t & 1];          // t=0 reads zeroed h0buf
    float*       hout = hb[(t + 1) & 1];    // t=1023 writes h0buf (final)

    v8f acc[4] = {};
    const float* aptr = hin + (size_t)arow * HID + khalf;
    for (int k0 = 0; k0 < HID; k0 += 32) {
      v16bf a = load_frag_f32(aptr + k0);
#pragma unroll
      for (int j = 0; j < 4; ++j) {
        const __bf16* bp = wlds + (size_t)(n0l + j * 16 + (lane & 15)) * HID + k0 + khalf;
        v16bf b = load_frag_bf16(bp);       // ds_load_b128 x2
        acc[j] = __builtin_amdgcn_wmma_f32_16x16x32_bf16(
            false, a, false, b, (short)0, acc[j], false, false);
      }
    }

    const float* xpt = xproj + (size_t)t * (BATCH * HID);
#pragma unroll
    for (int j = 0; j < 4; ++j) {
      const int col = n0g + j * 16 + (lane & 15);
#pragma unroll
      for (int r = 0; r < 8; ++r) {
        const int row = m0 + r + hi8;
        float xp = __builtin_nontemporal_load(&xpt[(size_t)row * HID + col]);
        hout[(size_t)row * HID + col] = tanhf(acc[j][r] + xp);
      }
    }

    // ---- grid-wide barrier (16 WGs, monotone counter, reset each launch) ----
    __threadfence();            // release h stores to device scope
    __syncthreads();
    if (threadIdx.x == 0) {
      __hip_atomic_fetch_add(cnt, 1, __ATOMIC_RELEASE, __HIP_MEMORY_SCOPE_AGENT);
      const int target = NWG2 * (t + 1);
      while (__hip_atomic_load(cnt, __ATOMIC_ACQUIRE, __HIP_MEMORY_SCOPE_AGENT) < target) {
        __builtin_amdgcn_s_sleep(2);
      }
    }
    __syncthreads();
    __threadfence();            // acquire other WGs' h stores
  }
}

// ---------------------------------------------------------------------------
// Kernel 3: final h (in h0buf after 1024 steps) -> d_out
// ---------------------------------------------------------------------------
__global__ __launch_bounds__(256) void rnn_copyout_kernel(
    const float* __restrict__ hfin, float* __restrict__ out) {
  int idx = blockIdx.x * 256 + threadIdx.x;
  if (idx < BATCH * HID) out[idx] = hfin[idx];
}

extern "C" void kernel_launch(void* const* d_in, const int* in_sizes, int n_in,
                              void* d_out, int out_size, void* d_ws, size_t ws_size,
                              hipStream_t stream) {
  (void)in_sizes; (void)n_in; (void)out_size; (void)ws_size;
  const float* x   = (const float*)d_in[0];  // [1024,256,512]
  const float* Wih = (const float*)d_in[1];  // [512,512]
  const float* Whh = (const float*)d_in[2];  // [512,512]
  const float* bih = (const float*)d_in[3];  // [512]
  const float* bhh = (const float*)d_in[4];  // [512]
  float* out = (float*)d_out;                // [256,512]

  char* ws = (char*)d_ws;
  __bf16* wih_bf = (__bf16*)(ws + OFF_WIH);
  __bf16* whh_bf = (__bf16*)(ws + OFF_WHH);
  float*  bias   = (float*)(ws + OFF_BIAS);
  float*  h0buf  = (float*)(ws + OFF_H0);
  float*  h1buf  = (float*)(ws + OFF_H1);
  int*    cnt    = (int*)(ws + OFF_CNT);
  float*  xproj  = (float*)(ws + OFF_XP);

  // 0) prep: bf16 weights, fused bias, h0 = 0, barrier counter = 0
  rnn_init_kernel<<<(DIM * HID + 255) / 256, 256, 0, stream>>>(
      Wih, Whh, bih, bhh, wih_bf, whh_bf, bias, h0buf, cnt);

  // 1) x_proj GEMM: 262144x512x512 -> 65536 waves (16M x 128N per wave,
  //    WG = 2 m-tiles x 4 n-groups for WGP$-shared A tiles)
  rnn_xproj_kernel<<<65536 / 8, 256, 0, stream>>>(x, wih_bf, bias, xproj);

  // 2) persistent recurrence: 4x4 WG grid, W_hh slab in 128KB dynamic LDS
  rnn_scan_kernel<<<NWG2, 256, 128 * HID * 2, stream>>>(
      xproj, whh_bf, h0buf, h1buf, cnt);

  // 3) copy final hidden state
  rnn_copyout_kernel<<<(BATCH * HID + 255) / 256, 256, 0, stream>>>(h0buf, out);
}
